// IsolationGNN_21921513079430
// MI455X (gfx1250) — matrix operand order repaired and checked by reading, not compiled, not run on hardware
//
#include <hip/hip_runtime.h>
#include <math.h>

// Problem constants (match reference)
#define N_NODES  100000
#define N_EDGES  3200000
#define F_NODE_  5
#define F_EDGE_  2
#define H        32
#define L_LAYERS 18

typedef __attribute__((ext_vector_type(16))) _Float16 v16h;
typedef __attribute__((ext_vector_type(8)))  _Float16 v8h;
typedef __attribute__((ext_vector_type(8)))  float    v8f;

// Pre-packed B-fragment table (f16, exact wave32 WMMA B layout):
//   frag = 32 lanes x 16 f16 = 512 f16 = 1 KB
//   ln_w: 2 frags/layer | lu_w: 4 frags/layer | c1_w: 2 frags
#define FRAG_ELEMS 512
#define FRAG_LN(l, nh)      ((l) * 2 + (nh))
#define FRAG_LU(l, kc, nh)  (L_LAYERS * 2 + (l) * 4 + (kc) * 2 + (nh))
#define FRAG_C1(nh)         (L_LAYERS * 6 + (nh))
#define N_FRAGS             (L_LAYERS * 6 + 2)   // 110

static __device__ __forceinline__ v8f wmma_f16(v16h a, v16h b, v8f c) {
    // D = A(16x32 f16) * B(32x16 f16) + C(16x16 f32)
    return __builtin_amdgcn_wmma_f32_16x16x32_f16(
        /*neg_a=*/false, a, /*neg_b=*/false, b,
        /*c_mod=*/(short)0, c, /*reuse_a=*/false, /*reuse_b=*/false);
}

// f16 row (32 f16, 64B) -> 16-bit A-matrix layout for one lane:
//   lanes 0-15:  elems 0-7 -> K 0-7,  elems 8-15 -> K 16-23
//   lanes 16-31: elems 0-7 -> K 8-15, elems 8-15 -> K 24-31
// Two contiguous 16B chunks -> 2x global_load_b128, zero conversions.
static __device__ __forceinline__ v16h load_a_row_f16(const _Float16* __restrict__ row, int lh) {
    const v8h* r8 = (const v8h*)row;
    v8h lo = r8[lh];
    v8h hi = r8[2 + lh];
    v16h a;
#pragma unroll
    for (int i = 0; i < 8; ++i) { a[i] = lo[i]; a[8 + i] = hi[i]; }
    return a;
}

// f32 row variant (for agg): 4x global_load_b128 + 8x v_cvt_pk_f16_f32.
static __device__ __forceinline__ v16h cvt_a_row_f32(float4 p0, float4 p1, float4 p2, float4 p3) {
    v16h a;
    a[0]  = (_Float16)p0.x; a[1]  = (_Float16)p0.y; a[2]  = (_Float16)p0.z; a[3]  = (_Float16)p0.w;
    a[4]  = (_Float16)p1.x; a[5]  = (_Float16)p1.y; a[6]  = (_Float16)p1.z; a[7]  = (_Float16)p1.w;
    a[8]  = (_Float16)p2.x; a[9]  = (_Float16)p2.y; a[10] = (_Float16)p2.z; a[11] = (_Float16)p2.w;
    a[12] = (_Float16)p3.x; a[13] = (_Float16)p3.y; a[14] = (_Float16)p3.z; a[15] = (_Float16)p3.w;
    return a;
}

// Load a pre-packed f16 B fragment (32B per lane -> 2x global_load_b128).
static __device__ __forceinline__ v16h load_b_frag(const _Float16* __restrict__ frag, int lane) {
    return *(const v16h*)(frag + lane * 16);
}

// ---------------------------------------------------------------------------
// Node encoder: h16 = f16(x @ Wn + bn)  ([N,5]@[5,32]) — K too small for WMMA.
// Grid exactly N*H threads (divides 256).
// ---------------------------------------------------------------------------
__global__ __launch_bounds__(256) void node_encode_kernel(
    const float* __restrict__ x, const float* __restrict__ Wn,
    const float* __restrict__ bn, _Float16* __restrict__ h16)
{
    int idx = blockIdx.x * blockDim.x + threadIdx.x;   // exact fit
    int i = idx / H, n = idx % H;
    float s = bn[n];
#pragma unroll
    for (int f = 0; f < F_NODE_; ++f) s += x[i * F_NODE_ + f] * Wn[f * H + n];
    h16[idx] = (_Float16)s;
}

// ---------------------------------------------------------------------------
// Fused per-layer edge weights:
//   Wf[l] = We @ le_w[l]  (2x32),  bf[l] = be @ le_w[l] + le_b[l]  (32)
// so ee = edge_attr @ Wf[l] + bf[l]  (never materialize ea[E,32] = 410 MB).
// Grid exactly L*H = 576 threads (3 x 192).
// ---------------------------------------------------------------------------
__global__ __launch_bounds__(192) void fuse_edge_weights_kernel(
    const float* __restrict__ We, const float* __restrict__ be,
    const float* __restrict__ le_w, const float* __restrict__ le_b,
    float* __restrict__ Wf, float* __restrict__ bf)
{
    int idx = blockIdx.x * blockDim.x + threadIdx.x;   // exact fit
    int l = idx / H, n = idx % H;
    const float* lw = le_w + l * H * H;
    float w0 = 0.f, w1 = 0.f, bb = 0.f;
#pragma unroll 4
    for (int k = 0; k < H; ++k) {
        float v = lw[k * H + n];
        w0 += We[0 * H + k] * v;
        w1 += We[1 * H + k] * v;
        bb += be[k] * v;
    }
    Wf[l * 2 * H + 0 * H + n] = w0;
    Wf[l * 2 * H + 1 * H + n] = w1;
    bf[l * H + n] = bb + le_b[l * H + n];
}

// ---------------------------------------------------------------------------
// Pre-pack all weight matrices into f16 WMMA B fragments (done once):
//   frag element i of lane L holds B[(L>>4)*8 + i + (i>=8 ? 8 : 0)][(L&15) + nh*16]
// Grid exactly N_FRAGS*32 = 3520 threads (22 x 160).
// ---------------------------------------------------------------------------
__global__ __launch_bounds__(160) void pack_b_frags_kernel(
    const float* __restrict__ ln_w,   // [L][H][H]
    const float* __restrict__ lu_w,   // [L][2H][H]
    const float* __restrict__ c1w,    // [H][H]
    _Float16* __restrict__ Bpack)
{
    int t = blockIdx.x * blockDim.x + threadIdx.x;     // exact fit
    int frag = t >> 5;
    int lane = t & 31;
    int kb = (lane >> 4) * 8;

    const float* srcm;
    int nh, krow_off;
    if (frag < L_LAYERS * 2) {                       // ln_w
        int l = frag >> 1; nh = frag & 1;
        srcm = ln_w + l * H * H; krow_off = 0;
    } else if (frag < L_LAYERS * 6) {                // lu_w
        int f = frag - L_LAYERS * 2;
        int l = f >> 2; int kc = (f >> 1) & 1; nh = f & 1;
        srcm = lu_w + l * 2 * H * H; krow_off = kc * H;
    } else {                                         // c1_w
        nh = frag - L_LAYERS * 6;
        srcm = c1w; krow_off = 0;
    }
    int n = (lane & 15) + nh * 16;
    _Float16* dstf = Bpack + frag * FRAG_ELEMS + lane * 16;
#pragma unroll
    for (int i = 0; i < 16; ++i)
        dstf[i] = (_Float16)srcm[(krow_off + kb + i + (i >= 8 ? 8 : 0)) * H + n];
}

// ---------------------------------------------------------------------------
// One-time zero of agg (float4-wide, exact fit: 800000 = 3125 x 256).
// ---------------------------------------------------------------------------
__global__ __launch_bounds__(256) void zero_kernel(float4* __restrict__ p)
{
    int idx = blockIdx.x * blockDim.x + threadIdx.x;
    p[idx] = make_float4(0.f, 0.f, 0.f, 0.f);
}

// ---------------------------------------------------------------------------
// Edge kernel (one wave == 16 edges, E exact multiple of 16*8 -> no guards):
//   t   = h[src] @ ln_w            (2x V_WMMA_F32_16X16X32_F16)
//   ee  = edge_attr @ Wf + bf      (fused, 2 FMAs/elem)
//   msg = xj*(1-ee) + ee*(t+ln_b)
//   agg[dst] += msg                (global_atomic_add_f32, L2-resident)
// ---------------------------------------------------------------------------
__global__ __launch_bounds__(256) void edge_msg_kernel(
    const _Float16* __restrict__ h16,       // [N][H] f16
    const float2* __restrict__ edge_attr,   // [E] pairs
    const int*   __restrict__ src,
    const int*   __restrict__ dst,
    const _Float16* __restrict__ blnpack,   // 2 packed frags for this layer
    const float* __restrict__ lnb,          // [H]
    const float* __restrict__ Wf,           // [2][H] fused, this layer
    const float* __restrict__ bf,           // [H]
    float* __restrict__ agg)
{
    const int wave  = blockIdx.x * 8 + (threadIdx.x >> 5);
    const int lane  = threadIdx.x & 31;
    const int lh    = lane >> 4;
    const int lr    = lane & 15;
    const int ebase = wave * 16;

    // ---- A: gathered source-node rows, f16 direct (2x b128, no cvt)
    const int srcA = src[ebase + lr];
    v16h a = load_a_row_f16(h16 + srcA * H, lh);

    // ---- B: pre-packed f16 fragments
    v16h b0 = load_b_frag(blnpack, lane);
    v16h b1 = load_b_frag(blnpack + FRAG_ELEMS, lane);

    v8f c0 = {}, c1 = {};
    c0 = wmma_f16(a, b0, c0);
    c1 = wmma_f16(a, b1, c1);

    // ---- loop-invariant per-n constants (n = lr and lr+16)
    const float wf0a = Wf[lr],      wf1a = Wf[H + lr],      bfa = bf[lr],      lnba = lnb[lr];
    const float wf0b = Wf[lr + 16], wf1b = Wf[H + lr + 16], bfb = bf[lr + 16], lnbb = lnb[lr + 16];

    // ---- elementwise mix + scatter-add (D layout: m = r + 8*lh, n = lr)
#pragma unroll
    for (int r = 0; r < 8; ++r) {
        const int e      = ebase + r + 8 * lh;
        const float2 eav = edge_attr[e];
        const int   s    = src[e];
        const int   d    = dst[e];
        const float xv0  = (float)h16[s * H + lr];
        const float xv1  = (float)h16[s * H + lr + 16];
        {
            float ee  = eav.x * wf0a + eav.y * wf1a + bfa;
            float msg = xv0 * (1.0f - ee) + ee * (c0[r] + lnba);
            unsafeAtomicAdd(&agg[d * H + lr], msg);
        }
        {
            float ee  = eav.x * wf0b + eav.y * wf1b + bfb;
            float msg = xv1 * (1.0f - ee) + ee * (c1[r] + lnbb);
            unsafeAtomicAdd(&agg[d * H + lr + 16], msg);
        }
    }
}

// ---------------------------------------------------------------------------
// Node update: hout16 = f16(relu(concat(h, agg) @ lu_w + lu_b))
// [N,64]@[64,32]: 2 K-chunks x 2 N-halves = 4 WMMAs per 16-node tile.
// Also re-zeroes agg in place for the next layer (each element touched once).
// Grid: 1250 blocks x 160 threads = 6250 waves = N/16 tiles exactly, no guards.
// ---------------------------------------------------------------------------
__global__ __launch_bounds__(160) void node_update_kernel(
    const _Float16* __restrict__ h16,
    float* __restrict__ agg,
    const _Float16* __restrict__ blupack,   // 4 packed frags for this layer
    const float* __restrict__ lub,          // [H]
    _Float16* __restrict__ hout16)
{
    const int wave  = blockIdx.x * 5 + (threadIdx.x >> 5);
    const int lane  = threadIdx.x & 31;
    const int lh    = lane >> 4;
    const int lr    = lane & 15;
    const int nbase = wave * 16;
    const int rowA  = nbase + lr;

    v16h a1 = load_a_row_f16(h16 + rowA * H, lh);

    float4* ar4 = (float4*)(agg + rowA * H);
    float4 q0 = ar4[lh * 2 + 0];
    float4 q1 = ar4[lh * 2 + 1];
    float4 q2 = ar4[lh * 2 + 4];
    float4 q3 = ar4[lh * 2 + 5];
    v16h a2 = cvt_a_row_f32(q0, q1, q2, q3);
    // re-zero agg for the next layer's scatter (lines are hot in L0)
    const float4 z = make_float4(0.f, 0.f, 0.f, 0.f);
    ar4[lh * 2 + 0] = z; ar4[lh * 2 + 1] = z;
    ar4[lh * 2 + 4] = z; ar4[lh * 2 + 5] = z;

    v16h b00 = load_b_frag(blupack + 0 * FRAG_ELEMS, lane);  // kc=0, nh=0
    v16h b01 = load_b_frag(blupack + 1 * FRAG_ELEMS, lane);  // kc=0, nh=1
    v16h b10 = load_b_frag(blupack + 2 * FRAG_ELEMS, lane);  // kc=1, nh=0
    v16h b11 = load_b_frag(blupack + 3 * FRAG_ELEMS, lane);  // kc=1, nh=1

    v8f c0 = {}, c1 = {};
    c0 = wmma_f16(a1, b00, c0);
    c0 = wmma_f16(a2, b10, c0);
    c1 = wmma_f16(a1, b01, c1);
    c1 = wmma_f16(a2, b11, c1);

    const float lb0 = lub[lr], lb1 = lub[lr + 16];
#pragma unroll
    for (int r = 0; r < 8; ++r) {
        const int node = nbase + r + 8 * lh;
        float v0 = c0[r] + lb0;
        float v1 = c1[r] + lb1;
        hout16[node * H + lr]      = (_Float16)(v0 > 0.f ? v0 : 0.f);
        hout16[node * H + lr + 16] = (_Float16)(v1 > 0.f ? v1 : 0.f);
    }
}

// ---------------------------------------------------------------------------
// Classifier stage 1: g = relu(h @ c1_w + c1_b)   (2 WMMAs per tile)
// Grid: 1250 x 160, exact fit.
// ---------------------------------------------------------------------------
__global__ __launch_bounds__(160) void classify1_kernel(
    const _Float16* __restrict__ h16,
    const _Float16* __restrict__ bc1pack,   // 2 packed frags
    const float* __restrict__ c1b,          // [H]
    float* __restrict__ g)
{
    const int wave  = blockIdx.x * 5 + (threadIdx.x >> 5);
    const int lane  = threadIdx.x & 31;
    const int lh    = lane >> 4;
    const int lr    = lane & 15;
    const int nbase = wave * 16;

    v16h a = load_a_row_f16(h16 + (nbase + lr) * H, lh);

    v16h b0 = load_b_frag(bc1pack, lane);
    v16h b1 = load_b_frag(bc1pack + FRAG_ELEMS, lane);

    v8f c0 = {}, c1 = {};
    c0 = wmma_f16(a, b0, c0);
    c1 = wmma_f16(a, b1, c1);

    const float cb0 = c1b[lr], cb1 = c1b[lr + 16];
#pragma unroll
    for (int r = 0; r < 8; ++r) {
        const int node = nbase + r + 8 * lh;
        float v0 = c0[r] + cb0;
        float v1 = c1[r] + cb1;
        g[node * H + lr]      = v0 > 0.f ? v0 : 0.f;
        g[node * H + lr + 16] = v1 > 0.f ? v1 : 0.f;
    }
}

// ---------------------------------------------------------------------------
// Classifier stage 2: out = sigmoid(g @ c2_w + c2_b)   ([H,1] dot, VALU)
// Grid: 625 x 160 = N exactly, no guard.
// ---------------------------------------------------------------------------
__global__ __launch_bounds__(160) void classify2_kernel(
    const float* __restrict__ g,
    const float* __restrict__ c2w, const float* __restrict__ c2b,
    float* __restrict__ out)
{
    int i = blockIdx.x * blockDim.x + threadIdx.x;     // exact fit
    const float4* g4 = (const float4*)(g + i * H);
    const float4* w4 = (const float4*)c2w;
    float s = c2b[0];
#pragma unroll
    for (int k = 0; k < H / 4; ++k) {
        float4 gv = g4[k], wv = w4[k];
        s += gv.x * wv.x + gv.y * wv.y + gv.z * wv.z + gv.w * wv.w;
    }
    out[i] = 1.0f / (1.0f + __expf(-s));
}

// ---------------------------------------------------------------------------
extern "C" void kernel_launch(void* const* d_in, const int* in_sizes, int n_in,
                              void* d_out, int out_size, void* d_ws, size_t ws_size,
                              hipStream_t stream) {
    const float* x         = (const float*)d_in[0];
    const float* edge_attr = (const float*)d_in[1];
    const int*   edge_idx  = (const int*)  d_in[2];
    const float* Wn        = (const float*)d_in[3];
    const float* bn        = (const float*)d_in[4];
    const float* We        = (const float*)d_in[5];
    const float* be        = (const float*)d_in[6];
    const float* ln_w      = (const float*)d_in[7];
    const float* ln_b      = (const float*)d_in[8];
    const float* le_w      = (const float*)d_in[9];
    const float* le_b      = (const float*)d_in[10];
    const float* lu_w      = (const float*)d_in[11];
    const float* lu_b      = (const float*)d_in[12];
    const float* c1w       = (const float*)d_in[13];
    const float* c1b       = (const float*)d_in[14];
    const float* c2w       = (const float*)d_in[15];
    const float* c2b       = (const float*)d_in[16];
    float* out = (float*)d_out;

    const int* src = edge_idx;
    const int* dst = edge_idx + N_EDGES;

    // Workspace (bytes): agg f32 | g (aliases agg) | h16A | h16B | Wf+bf | Bpack
    char* ws = (char*)d_ws;
    const size_t AGGB  = (size_t)N_NODES * H * sizeof(float);     // 12.8 MB
    const size_t H16B  = (size_t)N_NODES * H * sizeof(_Float16);  // 6.4 MB
    float*     agg   = (float*)(ws);
    _Float16*  h16A  = (_Float16*)(ws + AGGB);
    _Float16*  h16B  = (_Float16*)(ws + AGGB + H16B);
    float*     Wf    = (float*)(ws + AGGB + 2 * H16B);            // L*2*H floats
    float*     bf    = Wf + L_LAYERS * 2 * H;                     // L*H floats
    _Float16*  Bpack = (_Float16*)(ws + AGGB + 2 * H16B + 8192);  // 110 KB

    // --- one-time prep
    node_encode_kernel<<<(N_NODES * H) / 256, 256, 0, stream>>>(x, Wn, bn, h16A);
    fuse_edge_weights_kernel<<<3, 192, 0, stream>>>(We, be, le_w, le_b, Wf, bf);
    pack_b_frags_kernel<<<(N_FRAGS * 32) / 160, 160, 0, stream>>>(ln_w, lu_w, c1w, Bpack);
    zero_kernel<<<(N_NODES * H / 4) / 256, 256, 0, stream>>>((float4*)agg);

    // --- layers (node_update re-zeroes agg for the next layer)
    const int blocks_e = (N_EDGES / 16) / 8;   // 25000, exact
    const int blocks_n = (N_NODES / 16) / 5;   // 1250, exact

    _Float16* hin  = h16A;
    _Float16* hout = h16B;
    for (int l = 0; l < L_LAYERS; ++l) {
        edge_msg_kernel<<<blocks_e, 256, 0, stream>>>(
            hin, (const float2*)edge_attr, src, dst,
            Bpack + FRAG_LN(l, 0) * FRAG_ELEMS,
            ln_b + l * H, Wf + l * 2 * H, bf + l * H, agg);
        node_update_kernel<<<blocks_n, 160, 0, stream>>>(
            hin, agg, Bpack + FRAG_LU(l, 0, 0) * FRAG_ELEMS,
            lu_b + l * H, hout);
        _Float16* t = hin; hin = hout; hout = t;
    }

    // --- classifier (reuse agg as scratch for g; agg is zero, gets overwritten)
    classify1_kernel<<<blocks_n, 160, 0, stream>>>(
        hin, Bpack + FRAG_C1(0) * FRAG_ELEMS, c1b, (float*)agg);
    classify2_kernel<<<N_NODES / 160, 160, 0, stream>>>((float*)agg, c2w, c2b, out);
}
